// IntegratedLoss_26774826123526
// MI455X (gfx1250) — compile-verified, matrix operand order checked by reading, not run on hardware
//
#include <hip/hip_runtime.h>
#include <hip/hip_bf16.h>
#include <math.h>

#define DEV __device__ __forceinline__

// Problem constants (from reference)
constexpr int   BN = 2, AN = 16384, CN = 15, GN = 16;
constexpr int   BLK = 256;                 // 8 wave32 per block
constexpr int   ABLK = AN / BLK;           // 64 blocks per image
constexpr float ALPHA_F   = 0.25f;
constexpr float IOU_T     = 0.5f;
constexpr float BETA_F    = 1.0f / 9.0f;
constexpr float D2R       = 0.017453292519943295f;
constexpr float EPSF      = 1e-8f;
constexpr float WX = 10.f, WY = 10.f, WW = 10.f, WHT = 5.f, WT = 15.f;

typedef __attribute__((ext_vector_type(2)))  float    v2f;
typedef __attribute__((ext_vector_type(8)))  float    v8f;
typedef __attribute__((ext_vector_type(16))) _Float16 v16h;

// ---------------- wave32 reductions ----------------
DEV float wave_sum_f32(float v) {
    v += __shfl_xor(v, 16, 32);
    v += __shfl_xor(v, 8, 32);
    v += __shfl_xor(v, 4, 32);
    v += __shfl_xor(v, 2, 32);
    v += __shfl_xor(v, 1, 32);
    return v;
}

// Reduce TWO independent f32 quantities across the wave with ONE matrix op.
// A (16x4 f32): A[m,k] = 1 for (m<8, k even) and (m>=8, k odd), else 0.
// B (4x16 f32): VGPR0 carries rows K=0 (lanes 0-15) / K=2 (lanes 16-31) = cls,
//               VGPR1 carries rows K=1 / K=3 = reg.
// => D rows 0-7 hold cls(n)+cls(n+16), rows 8-15 hold reg(n)+reg(n+16).
// D.v0: lanes 0-15 = cls pair-sums, lanes 16-31 = reg pair-sums; 4 xor-shuffles
// (which never cross the 16-lane halves) finish both reductions in parallel.
DEV void wave_sum_pair(float cls_v, float reg_v, float& cls_tot, float& reg_tot,
                       int lane) {
#if __has_builtin(__builtin_amdgcn_wmma_f32_16x16x4_f32)
    const int m = lane & 15;
    v2f a; a[0] = (m < 8) ? 1.f : 0.f; a[1] = (m < 8) ? 0.f : 1.f;
    v2f bm; bm[0] = cls_v; bm[1] = reg_v;
    v8f c = {};
    c = __builtin_amdgcn_wmma_f32_16x16x4_f32(
            false, a, false, bm, (short)0, c, false, false);
    float w = c[0];
    w += __shfl_xor(w, 8, 32);
    w += __shfl_xor(w, 4, 32);
    w += __shfl_xor(w, 2, 32);
    w += __shfl_xor(w, 1, 32);
    cls_tot = __shfl(w, 0, 32);     // lanes 0-15 all hold cls total
    reg_tot = __shfl(w, 16, 32);    // lanes 16-31 all hold reg total
#else
    (void)lane;
    cls_tot = wave_sum_f32(cls_v);
    reg_tot = wave_sum_f32(reg_v);
#endif
}

// Reduce 32 per-lane {0,1} flags on the matrix pipe (exact in f16, f32 acc).
DEV float wave_sum_bin(float v) {
#if __has_builtin(__builtin_amdgcn_wmma_f32_16x16x32_f16)
    v16h a;
#pragma unroll
    for (int i = 0; i < 16; ++i) a[i] = (_Float16)1.0f;
    v16h bm = {};
    bm[0] = (_Float16)v;
    v8f c = {};
    c = __builtin_amdgcn_wmma_f32_16x16x32_f16(
            false, a, false, bm, (short)0, c, false, false);
    float w = c[0];               // lane n holds v(n%16) + v(n%16 + 16)
    w += __shfl_xor(w, 8, 32);
    w += __shfl_xor(w, 4, 32);
    w += __shfl_xor(w, 2, 32);
    w += __shfl_xor(w, 1, 32);
    return w;
#else
    return wave_sum_f32(v);
#endif
}

// ---------------- geometry helpers ----------------
DEV bool pin_rect(float px, float py, float cx, float cy, float w, float h,
                  float c, float s) {
    float dx = px - cx, dy = py - cy;
    float lx = dx * c + dy * s;
    float ly = -dx * s + dy * c;
    return (fabsf(lx) <= 0.5f * w + 1e-4f) && (fabsf(ly) <= 0.5f * h + 1e-4f);
}

DEV float rot_iou(float ax, float ay, float aw, float ah, float ac, float as,
                  const float* acx, const float* acy,
                  float gx, float gy, float gw, float gh, float gc, float gs,
                  const float* gcx, const float* gcy) {
    float px[24], py[24];
    int cnt = 0;
#pragma unroll
    for (int i = 0; i < 4; ++i)
        if (pin_rect(acx[i], acy[i], gx, gy, gw, gh, gc, gs)) {
            px[cnt] = acx[i]; py[cnt] = acy[i]; ++cnt;
        }
#pragma unroll
    for (int j = 0; j < 4; ++j)
        if (pin_rect(gcx[j], gcy[j], ax, ay, aw, ah, ac, as)) {
            px[cnt] = gcx[j]; py[cnt] = gcy[j]; ++cnt;
        }
#pragma unroll
    for (int i = 0; i < 4; ++i) {
        int i2 = (i + 1) & 3;
        float p0x = acx[i], p0y = acy[i];
        float d1x = acx[i2] - p0x, d1y = acy[i2] - p0y;
#pragma unroll
        for (int j = 0; j < 4; ++j) {
            int j2 = (j + 1) & 3;
            float q0x = gcx[j], q0y = gcy[j];
            float d2x = gcx[j2] - q0x, d2y = gcy[j2] - q0y;
            float rx = q0x - p0x, ry = q0y - p0y;
            float den = d1x * d2y - d1y * d2x;
            if (fabsf(den) >= EPSF) {
                float tt = (rx * d2y - ry * d2x) / den;
                float uu = (rx * d1y - ry * d1x) / den;
                if (tt >= 0.f && tt <= 1.f && uu >= 0.f && uu <= 1.f) {
                    px[cnt] = p0x + tt * d1x; py[cnt] = p0y + tt * d1y; ++cnt;
                }
            }
        }
    }
    float inter = 0.f;
    if (cnt >= 3) {
        float cx = 0.f, cy = 0.f;
        for (int i = 0; i < cnt; ++i) { cx += px[i]; cy += py[i]; }
        float inv = 1.0f / (float)cnt;
        cx *= inv; cy *= inv;
        float ang[24];
        for (int i = 0; i < cnt; ++i) {
            px[i] -= cx; py[i] -= cy;
            ang[i] = atan2f(py[i], px[i]);
        }
        for (int i = 1; i < cnt; ++i) {   // insertion sort by angle
            float ka = ang[i], kx = px[i], ky = py[i];
            int j = i - 1;
            while (j >= 0 && ang[j] > ka) {
                ang[j + 1] = ang[j]; px[j + 1] = px[j]; py[j + 1] = py[j]; --j;
            }
            ang[j + 1] = ka; px[j + 1] = kx; py[j + 1] = ky;
        }
        float area = 0.f;
        for (int i = 0; i < cnt; ++i) {
            int j = (i + 1 == cnt) ? 0 : i + 1;
            area += px[i] * py[j] - py[i] * px[j];
        }
        inter = 0.5f * fabsf(area);
    }
    float uni = aw * ah + gw * gh - inter;
    return inter / fmaxf(uni, EPSF);
}

// ---------------- kernels ----------------
__global__ void k0_init(unsigned long long* keys) {
    int t = threadIdx.x;
    if (t < BN * GN) keys[t] = 0ull;
}

__global__ __launch_bounds__(BLK) void k1_pairs(
        const float* __restrict__ anch, const float* __restrict__ ann,
        float* __restrict__ iouMax, int* __restrict__ iouArg,
        unsigned long long* __restrict__ keys) {
    __shared__ float gX[GN], gY[GN], gW[GN], gH[GN], gC[GN], gS[GN], gLab[GN];
    __shared__ float gCx[GN][4], gCy[GN][4];
    __shared__ unsigned long long skey[GN];
    const int t = threadIdx.x;
    const int b = blockIdx.x / ABLK;
    const int a = (blockIdx.x % ABLK) * BLK + t;

    if (t < GN) {
        const float* g = ann + (size_t)(b * GN + t) * 6;
        float x = g[0], y = g[1], w = g[2], h = g[3], th = g[4] * D2R;
        float c = cosf(th), s = sinf(th);
        gX[t] = x; gY[t] = y; gW[t] = w; gH[t] = h; gC[t] = c; gS[t] = s;
        gLab[t] = g[5];
        const float sx[4] = {-1.f, 1.f, 1.f, -1.f};
        const float sy[4] = {-1.f, -1.f, 1.f, 1.f};
#pragma unroll
        for (int i = 0; i < 4; ++i) {
            float dx = 0.5f * w * sx[i], dy = 0.5f * h * sy[i];
            gCx[t][i] = x + dx * c - dy * s;
            gCy[t][i] = y + dx * s + dy * c;
        }
        skey[t] = 0ull;
    }
    __syncthreads();

    const float* e = anch + (size_t)(b * AN + a) * 5;
    float ax = e[0], ay = e[1], aw = e[2], ah = e[3], ath = e[4] * D2R;
    float ac = cosf(ath), asn = sinf(ath);
    float acx[4], acy[4];
    {
        const float sx[4] = {-1.f, 1.f, 1.f, -1.f};
        const float sy[4] = {-1.f, -1.f, 1.f, 1.f};
#pragma unroll
        for (int i = 0; i < 4; ++i) {
            float dx = 0.5f * aw * sx[i], dy = 0.5f * ah * sy[i];
            acx[i] = ax + dx * ac - dy * asn;
            acy[i] = ay + dx * asn + dy * ac;
        }
    }
    float sa = 0.5f * fmaxf(aw, ah);
    float best = -2.0f;
    int besti = 0;
    for (int g = 0; g < GN; ++g) {
        float val;
        if (gLab[g] == -1.0f) {
            val = -1.0f;
        } else {
            float sb = 0.5f * fmaxf(gW[g], gH[g]);
            float iw = fmaxf(fminf(ax + sa, gX[g] + sb) - fmaxf(ax - sa, gX[g] - sb), 0.f);
            float ih = fmaxf(fminf(ay + sa, gY[g] + sb) - fmaxf(ay - sa, gY[g] - sb), 0.f);
            float inter = iw * ih;
            float uni = 4.f * sa * sa + 4.f * sb * sb - inter;
            float ind = inter / fmaxf(uni, EPSF);
            val = (ind > 0.1f)
                      ? rot_iou(ax, ay, aw, ah, ac, asn, acx, acy,
                                gX[g], gY[g], gW[g], gH[g], gC[g], gS[g],
                                &gCx[g][0], &gCy[g][0])
                      : 0.f;
        }
        if (val > best) { best = val; besti = g; }   // first-index argmax
        // pack (monotone float, ~a) so u64 max == (max value, then min index)
        unsigned vb = __float_as_uint(val);
        vb = (vb & 0x80000000u) ? ~vb : (vb | 0x80000000u);
        unsigned long long key =
            ((unsigned long long)vb << 32) | (unsigned)(~(unsigned)a);
        atomicMax(&skey[g], key);
    }
    iouMax[b * AN + a] = best;
    iouArg[b * AN + a] = besti;
    __syncthreads();
    if (t < GN) atomicMax(&keys[b * GN + t], skey[t]);
}

__global__ void k2_force(const float* __restrict__ ann,
                         const unsigned long long* __restrict__ keys,
                         int* __restrict__ forced, int* __restrict__ hasgt) {
    int t = threadIdx.x;
    if (t < BN * GN) {
        int b = t / GN, g = t % GN;
        float lab = ann[(size_t)(b * GN + g) * 6 + 5];
        unsigned long long k = keys[t];
        unsigned vb = (unsigned)(k >> 32);
        unsigned ob = (vb & 0x80000000u) ? (vb & 0x7fffffffu) : ~vb;
        float mg = __uint_as_float(ob);
        unsigned arg = ~(unsigned)(k & 0xffffffffull);
        bool force = (lab != -1.0f) && (mg < IOU_T);
        forced[t] = force ? (int)arg : -1;
    }
    if (t < BN) {
        int any = 0;
        for (int g = 0; g < GN; ++g)
            if (ann[(size_t)(t * GN + g) * 6 + 5] != -1.0f) any = 1;
        hasgt[t] = any;
    }
}

__global__ __launch_bounds__(BLK) void k3_loss(
        const float* __restrict__ clsP, const float* __restrict__ regP,
        const float* __restrict__ anch, const float* __restrict__ ann,
        const float* __restrict__ iouMax, const int* __restrict__ iouArg,
        const int* __restrict__ forced, float* __restrict__ part) {
    __shared__ float annL[GN][6];
    __shared__ int   fL[GN];
    __shared__ float wsum[BLK / 32][3];
    const int t = threadIdx.x;
    const int b = blockIdx.x / ABLK;
    const int a = (blockIdx.x % ABLK) * BLK + t;
    if (t < GN) {
#pragma unroll
        for (int k = 0; k < 6; ++k) annL[t][k] = ann[(size_t)(b * GN + t) * 6 + k];
        fL[t] = forced[b * GN + t];
    }
    __syncthreads();

    const size_t idx = (size_t)b * AN + a;
    __builtin_prefetch(clsP + idx * CN, 0, 0);   // global_prefetch_b8
    __builtin_prefetch(regP + idx * 5, 0, 0);

    float imax = iouMax[idx];
    int   iarg = iouArg[idx];
    bool pos = (imax >= IOU_T);
#pragma unroll
    for (int g = 0; g < GN; ++g) pos = pos || (fL[g] == a);

    float cls_c = 0.f, reg_c = 0.f;
    const float pos_c = pos ? 1.f : 0.f;
    const int alab = (int)annL[iarg][5];
    const int mode = pos ? 2 : ((imax < IOU_T - 0.1f) ? 1 : 0);
    if (mode) {
        const float* pc = clsP + idx * CN;
        for (int c = 0; c < CN; ++c) {
            float p = fminf(fmaxf(pc[c], 1e-4f), 1.f - 1e-4f);
            bool is1 = (mode == 2) && (c == alab);
            float af = is1 ? ALPHA_F : (1.f - ALPHA_F);
            float pt = is1 ? (1.f - p) : p;
            float fw = af * pt * pt;                       // gamma == 2
            float bce = is1 ? -logf(p + 1e-6f) : -logf(1.f - p + 1e-6f);
            cls_c += fw * bce;
        }
    }
    if (pos) {
        const float* e = anch + idx * 5;
        const float* r = regP + idx * 5;
        float ew = fmaxf(e[2], 1.f), eh = fmaxf(e[3], 1.f);
        float gw = fmaxf(annL[iarg][2], 1.f), gh = fmaxf(annL[iarg][3], 1.f);
        float tgt[5];
        tgt[0] = WX * (annL[iarg][0] - e[0]) / ew;
        tgt[1] = WY * (annL[iarg][1] - e[1]) / eh;
        tgt[2] = WW * logf(gw / ew);
        tgt[3] = WHT * logf(gh / eh);
        tgt[4] = WT * (tanf(annL[iarg][4] * D2R) - tanf(e[4] * D2R));
#pragma unroll
        for (int k = 0; k < 5; ++k) {
            float d = fabsf(r[k] - tgt[k]);
            reg_c += (d < BETA_F) ? 0.5f * d * d / BETA_F : d - 0.5f * BETA_F;
        }
    }

    const int wid = t >> 5, lane = t & 31;
    float cs, rs;
    wave_sum_pair(cls_c, reg_c, cs, rs, lane);   // one v_wmma, both f32 sums
    float ps = wave_sum_bin(pos_c);              // matrix-pipe 0/1 reduction
    if (lane == 0) { wsum[wid][0] = cs; wsum[wid][1] = rs; wsum[wid][2] = ps; }
    __syncthreads();
    if (t == 0) {
        float c0 = 0.f, r0 = 0.f, p0 = 0.f;
        for (int w = 0; w < BLK / 32; ++w) {
            c0 += wsum[w][0]; r0 += wsum[w][1]; p0 += wsum[w][2];
        }
        part[blockIdx.x * 3 + 0] = c0;
        part[blockIdx.x * 3 + 1] = r0;
        part[blockIdx.x * 3 + 2] = p0;
    }
}

__global__ void k4_final(const float* __restrict__ part,
                         const int* __restrict__ hasgt,
                         float* __restrict__ out) {
    if (threadIdx.x == 0 && blockIdx.x == 0) {
        float clsm = 0.f, regm = 0.f;
        for (int b = 0; b < BN; ++b) {
            float cs = 0.f, rs = 0.f, np = 0.f;
            for (int blk = 0; blk < ABLK; ++blk) {
                int i = (b * ABLK + blk) * 3;
                cs += part[i + 0]; rs += part[i + 1]; np += part[i + 2];
            }
            float cl = cs / fmaxf(np, 1.f);
            float rl = (np > 0.f) ? rs / fmaxf(np * 5.f, 1.f) : 0.f;
            if (!hasgt[b]) { cl = 0.f; rl = 0.f; }
            clsm += cl; regm += rl;
        }
        out[0] = clsm / (float)BN;
        out[1] = regm / (float)BN;
    }
}

extern "C" void kernel_launch(void* const* d_in, const int* in_sizes, int n_in,
                              void* d_out, int out_size, void* d_ws, size_t ws_size,
                              hipStream_t stream) {
    (void)in_sizes; (void)n_in; (void)out_size; (void)ws_size;
    const float* clsP = (const float*)d_in[0];
    const float* regP = (const float*)d_in[1];
    const float* anch = (const float*)d_in[2];
    const float* ann  = (const float*)d_in[3];
    float* out = (float*)d_out;

    char* w = (char*)d_ws;
    float* iouMax = (float*)w;                         w += sizeof(float) * BN * AN;
    int* iouArg = (int*)w;                             w += sizeof(int) * BN * AN;
    unsigned long long* keys = (unsigned long long*)w; w += sizeof(unsigned long long) * BN * GN;
    int* forced = (int*)w;                             w += sizeof(int) * BN * GN;
    int* hasgt = (int*)w;                              w += sizeof(int) * BN;
    float* part = (float*)w;                           // BN*ABLK*3 floats

    const int blocks = BN * ABLK;
    k0_init<<<1, 64, 0, stream>>>(keys);
    k1_pairs<<<blocks, BLK, 0, stream>>>(anch, ann, iouMax, iouArg, keys);
    k2_force<<<1, 64, 0, stream>>>(ann, keys, forced, hasgt);
    k3_loss<<<blocks, BLK, 0, stream>>>(clsP, regP, anch, ann, iouMax, iouArg,
                                        forced, part);
    k4_final<<<1, 32, 0, stream>>>(part, hasgt, out);
}
